// EwaldModel_3238405341273
// MI455X (gfx1250) — matrix operand order repaired and checked by reading, not compiled, not run on hardware
//
#include <hip/hip_runtime.h>

// CDNA5 / gfx1250: wave32, WMMA 16x16x32 f16 -> f32.
typedef __attribute__((ext_vector_type(16))) _Float16 v16h;
typedef __attribute__((ext_vector_type(8)))  _Float16 v8h;
typedef __attribute__((ext_vector_type(8)))  float    v8f;

#define RES 2048
#define NB  256
#define K0F 50.0f

// ---------------------------------------------------------------------------
// Phase 1: build separable per-beam factor matrices (f16, [pos][beam], K
// contiguous so WMMA fragment loads are wide contiguous vector loads).
//   x-side:  Axt = re*u - im*v,  Bxt = -(re*v + im*u),  u=ax*cos(kx*x), v=ax*sin(kx*x)
//   y-side:  Pt  = ay*cos(ky*y), Qt  = ay*sin(ky*y)
// Then E_re = A^T P + B^T Q,  E_im = A^T Q - B^T P  (rank-256 each).
// ---------------------------------------------------------------------------
__global__ void ewald_factors(const float* __restrict__ theta,
                              const float* __restrict__ re_,
                              const float* __restrict__ im_,
                              const float* __restrict__ x0,
                              const float* __restrict__ y0,
                              const float* __restrict__ ls,
                              const float* __restrict__ X,
                              const float* __restrict__ Y,
                              _Float16* __restrict__ Axt,
                              _Float16* __restrict__ Bxt,
                              _Float16* __restrict__ Pt,
                              _Float16* __restrict__ Qt)
{
    const int beam = threadIdx.x;           // 256 beams per block
    const int pos  = blockIdx.x & (RES - 1);
    const int axis = blockIdx.x >> 11;      // 0 = x-axis, 1 = y-axis

    const float th = theta[beam];
    const float r  = re_[beam];
    const float i  = im_[beam];
    const float sg = __expf(ls[beam]) + 0.001f;
    const float inv_s2 = 1.0f / (sg * sg);

    if (axis == 0) {
        const float xv = X[(size_t)pos * RES];   // X[ix][*] == x[ix] (indexing='ij')
        const float kx = K0F * __cosf(th);
        const float dx = xv - x0[beam];
        const float ax = __expf(-dx * dx * inv_s2);
        float s, c;
        __sincosf(kx * xv, &s, &c);
        const float u = ax * c;
        const float v = ax * s;
        Axt[pos * NB + beam] = (_Float16)(r * u - i * v);
        Bxt[pos * NB + beam] = (_Float16)(-(r * v + i * u));
    } else {
        const float yv = Y[pos];                 // Y[*][iy] == y[iy]
        const float ky = K0F * __sinf(th);
        const float dy = yv - y0[beam];
        const float ay = __expf(-dy * dy * inv_s2);
        float s, c;
        __sincosf(ky * yv, &s, &c);
        Pt[pos * NB + beam] = (_Float16)(ay * c);
        Qt[pos * NB + beam] = (_Float16)(ay * s);
    }
}

// Fragment loaders (layouts per CDNA5 ISA 7.12.2, wave32):
//   A-operand (16x32, MxK): lanes 0-15 row M=lane%16, halfs0-7 = K kb+8*kh..,
//                           halfs8-15 = K kb+16+8*kh..  (kh = lane/16)
//   B-operand (32x16, KxN): col N=lane%16, halfs0-7 = K kb+8*kh..,
//                           halfs8-15 = K kb+16+8*kh..
__device__ __forceinline__ v16h load_afrag(const _Float16* __restrict__ row,
                                           int kb, int kh) {
    union { v16h v; v8h h[2]; } u;
    u.h[0] = *(const v8h*)(row + kb + 8 * kh);
    u.h[1] = *(const v8h*)(row + kb + 16 + 8 * kh);
    return u.v;
}
__device__ __forceinline__ v16h load_bfrag(const _Float16* __restrict__ row,
                                           int kb, int kh) {
    union { v16h v; v8h h[2]; } u;
    u.h[0] = *(const v8h*)(row + kb + 8 * kh);
    u.h[1] = *(const v8h*)(row + kb + 16 + 8 * kh);
    return u.v;
}

// ---------------------------------------------------------------------------
// Phase 2: four fused rank-256 GEMMs via v_wmma_f32_16x16x32_f16.
// 2x2 register blocking: one wave32 computes a 32x32 output region
// (4 WMMA tiles, 16 f32 accumulator sets). Per k-step: 8 fragment loads
// feed 16 WMMAs (2x the compute:load ratio of 1-tile waves, half the L2
// read traffic, which is the dominant cost by roofline).
// ---------------------------------------------------------------------------
__global__ void __launch_bounds__(256)
ewald_gemm(const _Float16* __restrict__ Axt,
           const _Float16* __restrict__ Bxt,
           const _Float16* __restrict__ Pt,
           const _Float16* __restrict__ Qt,
           float* __restrict__ out)
{
    const int lane = threadIdx.x & 31;
    const int wid  = threadIdx.x >> 5;
    const int reg  = blockIdx.x * 8 + wid;       // 64x64 regions of 32x32
    const int tm   = (reg >> 6) << 5;            // output row base (x index)
    const int tn   = (reg & 63) << 5;            // output col base (y index)
    const int kh   = lane >> 4;                  // which 16-lane half of the wave
    const int l    = lane & 15;

    const _Float16* Arow[2] = { Axt + (tm + l) * NB,      Axt + (tm + 16 + l) * NB };
    const _Float16* Brow[2] = { Bxt + (tm + l) * NB,      Bxt + (tm + 16 + l) * NB };
    const _Float16* Prow[2] = { Pt  + (tn + l) * NB,      Pt  + (tn + 16 + l) * NB };
    const _Float16* Qrow[2] = { Qt  + (tn + l) * NB,      Qt  + (tn + 16 + l) * NB };

    v8f accAP[2][2] = {{{}, {}}, {{}, {}}};
    v8f accBQ[2][2] = {{{}, {}}, {{}, {}}};
    v8f accAQ[2][2] = {{{}, {}}, {{}, {}}};
    v8f accBP[2][2] = {{{}, {}}, {{}, {}}};

#pragma unroll
    for (int kb = 0; kb < NB; kb += 32) {
        v16h af[2], bf[2], pf[2], qf[2];
#pragma unroll
        for (int t = 0; t < 2; ++t) {
            af[t] = load_afrag(Arow[t], kb, kh);
            bf[t] = load_afrag(Brow[t], kb, kh);
            pf[t] = load_bfrag(Prow[t], kb, kh);
            qf[t] = load_bfrag(Qrow[t], kb, kh);
        }
#pragma unroll
        for (int mi = 0; mi < 2; ++mi) {
#pragma unroll
            for (int ni = 0; ni < 2; ++ni) {
                accAP[mi][ni] = __builtin_amdgcn_wmma_f32_16x16x32_f16(
                    false, af[mi], false, pf[ni], (short)0, accAP[mi][ni], false, false);
                accBQ[mi][ni] = __builtin_amdgcn_wmma_f32_16x16x32_f16(
                    false, bf[mi], false, qf[ni], (short)0, accBQ[mi][ni], false, false);
                accAQ[mi][ni] = __builtin_amdgcn_wmma_f32_16x16x32_f16(
                    false, af[mi], false, qf[ni], (short)0, accAQ[mi][ni], false, false);
                accBP[mi][ni] = __builtin_amdgcn_wmma_f32_16x16x32_f16(
                    false, bf[mi], false, pf[ni], (short)0, accBP[mi][ni], false, false);
            }
        }
    }

    // Epilogue: |E|^2 = (A^T P + B^T Q)^2 + (A^T Q - B^T P)^2
    // C/D layout: VGPR r -> row r (+8 for lanes 16-31), col = lane%16.
#pragma unroll
    for (int mi = 0; mi < 2; ++mi) {
#pragma unroll
        for (int ni = 0; ni < 2; ++ni) {
            float* orow = out + (size_t)(tm + 16 * mi + 8 * kh) * RES + tn + 16 * ni + l;
#pragma unroll
            for (int r = 0; r < 8; ++r) {
                const float er = accAP[mi][ni][r] + accBQ[mi][ni][r];
                const float ei = accAQ[mi][ni][r] - accBP[mi][ni][r];
                orow[(size_t)r * RES] = er * er + ei * ei;
            }
        }
    }
}

// ---------------------------------------------------------------------------
extern "C" void kernel_launch(void* const* d_in, const int* in_sizes, int n_in,
                              void* d_out, int out_size, void* d_ws, size_t ws_size,
                              hipStream_t stream) {
    const float* theta = (const float*)d_in[0];
    const float* re_   = (const float*)d_in[1];
    const float* im_   = (const float*)d_in[2];
    const float* x0    = (const float*)d_in[3];
    const float* y0    = (const float*)d_in[4];
    const float* ls    = (const float*)d_in[5];
    const float* X     = (const float*)d_in[6];
    const float* Y     = (const float*)d_in[7];
    float* out = (float*)d_out;

    // Workspace: 4 factor matrices, each 2048x256 f16 = 1 MB (4 MB total).
    char* ws = (char*)d_ws;
    const size_t arr = (size_t)RES * NB * sizeof(_Float16);
    _Float16* Axt = (_Float16*)(ws + 0 * arr);
    _Float16* Bxt = (_Float16*)(ws + 1 * arr);
    _Float16* Pt  = (_Float16*)(ws + 2 * arr);
    _Float16* Qt  = (_Float16*)(ws + 3 * arr);

    // Phase 1: 2 axes x 2048 positions, 256 beams per block.
    ewald_factors<<<2 * RES, NB, 0, stream>>>(theta, re_, im_, x0, y0, ls, X, Y,
                                              Axt, Bxt, Pt, Qt);
    // Phase 2: 64x64 regions of 32x32 output, 8 wave32 regions per block.
    ewald_gemm<<<(64 * 64) / 8, 256, 0, stream>>>(Axt, Bxt, Pt, Qt, out);
}
